// SelfAttention_v1_63161789055034
// MI455X (gfx1250) — matrix operand order repaired and checked by reading, not compiled, not run on hardware
//
#include <hip/hip_runtime.h>
#include <hip/hip_bf16.h>

#define SEQ 8192
#define EMB 768
#define DIM 64

typedef __attribute__((ext_vector_type(16))) _Float16 v16h;
typedef __attribute__((ext_vector_type(8)))  _Float16 v8h;
typedef __attribute__((ext_vector_type(8)))  float    v8f;
typedef __attribute__((ext_vector_type(4)))  float    v4f;
typedef __attribute__((ext_vector_type(4)))  unsigned int v4u;
typedef __attribute__((ext_vector_type(8)))  unsigned int v8u;

static __device__ __forceinline__ v8f wmma_f16(v16h a, v16h b, v8f c) {
    // D = A(16x32 f16) * B(32x16 f16) + C(16x16 f32)
    return __builtin_amdgcn_wmma_f32_16x16x32_f16(false, a, false, b,
                                                  (short)0, c, false, false);
}

// ---------------------------------------------------------------------------
// TDM: issue a 2D tensor_load_to_lds (2-SGPR-group form, <=2D tensor).
// Descriptor per CDNA5 ISA. data_size = 2 bytes (f16).
//   lds_off : LDS byte offset of tile destination
//   gaddr   : global address of TILE start
//   td0,td1 : tensor dims (elements);  t0,t1 : tile dims (elements)
//   s0      : tensor dim0 stride (elements)
// ---------------------------------------------------------------------------
static __device__ __forceinline__ void tdm_load_2d(unsigned lds_off,
                                                   const void* gaddr,
                                                   unsigned td0, unsigned td1,
                                                   unsigned t0, unsigned t1,
                                                   unsigned long long s0) {
    unsigned long long ga = (unsigned long long)(uintptr_t)gaddr;
    v4u g0;
    g0[0] = 1u;                                   // count=1, user mode, no gather
    g0[1] = lds_off;                              // lds_addr (bytes)
    g0[2] = (unsigned)(ga & 0xffffffffu);         // global_addr[31:0]
    g0[3] = (unsigned)(ga >> 32) | 0x80000000u;   // global_addr[56:32] | type=2
    v8u g1;
    g1[0] = 1u << 16;                             // data_size=1 -> 2-byte elems
    g1[1] = (td0 & 0xffffu) << 16;                // tensor_dim0[15:0]
    g1[2] = (td0 >> 16) | ((td1 & 0xffffu) << 16);// td0[31:16] | td1[15:0]
    g1[3] = (td1 >> 16) | (t0 << 16);             // td1[31:16] | tile_dim0
    g1[4] = t1;                                   // tile_dim1 (tile_dim2 = 0)
    g1[5] = (unsigned)(s0 & 0xffffffffu);         // dim0_stride[31:0]
    g1[6] = (unsigned)(s0 >> 32);                 // dim0_stride[47:32]
    g1[7] = 0u;
    asm volatile("tensor_load_to_lds %0, %1" :: "s"(g0), "s"(g1) : "memory");
}

// ---------------------------------------------------------------------------
// Kernel 1: transpose + convert W_Q/W_K/W_V [E,64] f32 -> Wt[3][64][768] f16
// ---------------------------------------------------------------------------
__global__ void attn_wt_kernel(const float* __restrict__ WQ,
                               const float* __restrict__ WK,
                               const float* __restrict__ WV,
                               _Float16* __restrict__ Wt) {
    int i = blockIdx.x * blockDim.x + threadIdx.x;
    if (i >= 3 * DIM * EMB) return;
    int m = i / (DIM * EMB);
    int r = i % (DIM * EMB);
    int n = r / EMB;
    int c = r % EMB;
    const float* W = (m == 0) ? WQ : (m == 1) ? WK : WV;
    Wt[i] = (_Float16)W[(size_t)c * DIM + n];
}

// ---------------------------------------------------------------------------
// Kernel 2: fused Q/K/V projection via WMMA. One wave computes the 16-row x
// 64-col tile of ALL THREE projections (A fragment loaded once, 12 WMMAs per
// 32-wide contraction step). Q,K row-major f16; V stored transposed.
// ---------------------------------------------------------------------------
__global__ __launch_bounds__(64) void attn_proj_kernel(
        const float* __restrict__ X, const _Float16* __restrict__ Wt,
        _Float16* __restrict__ Qh, _Float16* __restrict__ Kh,
        _Float16* __restrict__ Vt) {
    const int lane = threadIdx.x & 31;
    const int wave = threadIdx.x >> 5;
    const int tile = blockIdx.x * 2 + wave;
    const int r0   = tile * 16;
    const int n    = lane & 15;
    const int b    = lane >> 4;

    v8f acc[3][4];
    for (int m = 0; m < 3; ++m)
        for (int j = 0; j < 4; ++j)
            for (int v = 0; v < 8; ++v) acc[m][j][v] = 0.f;

    const float* xrow = X + (size_t)(r0 + n) * EMB;

    for (int c = 0; c < EMB; c += 32) {
        // A fragment: half h -> K = c + 16b + h
        v4f f0 = *(const v4f*)(xrow + c + 16 * b);
        v4f f1 = *(const v4f*)(xrow + c + 16 * b + 4);
        v4f f2 = *(const v4f*)(xrow + c + 16 * b + 8);
        v4f f3 = *(const v4f*)(xrow + c + 16 * b + 12);
        v16h a;
        for (int i = 0; i < 4; ++i) {
            a[i]      = (_Float16)f0[i];
            a[4 + i]  = (_Float16)f1[i];
            a[8 + i]  = (_Float16)f2[i];
            a[12 + i] = (_Float16)f3[i];
        }
        for (int m = 0; m < 3; ++m) {
            for (int j = 0; j < 4; ++j) {
                // B fragment: half h -> K = c + 16b + h (one 32B load)
                const _Float16* wp =
                    Wt + ((size_t)m * DIM + 16 * j + n) * EMB + c + 16 * b;
                v16h bf = *(const v16h*)wp;
                acc[m][j] = wmma_f16(a, bf, acc[m][j]);
            }
        }
    }

    // C layout: VGPR v, lane -> (M = v + 8*(lane/16), N = lane%16)
    for (int j = 0; j < 4; ++j) {
        for (int v = 0; v < 8; ++v) {
            int M = v + 8 * b;
            int col = 16 * j + n;
            Qh[(size_t)(r0 + M) * DIM + col] = (_Float16)acc[0][j][v];
            Kh[(size_t)(r0 + M) * DIM + col] = (_Float16)acc[1][j][v];
            Vt[(size_t)col * SEQ + r0 + M]   = (_Float16)acc[2][j][v];
        }
    }
}

// ---------------------------------------------------------------------------
// Kernel 3: flash attention, 4 waves/block, 16 query rows per wave.
// K/V tiles (32 keys) staged through double-buffered LDS by the Tensor Data
// Mover (wave0 issues, TENSORcnt + barrier handshake); all waves consume
// B-fragments from LDS. Online softmax with shuffle reductions + exp2.
// ---------------------------------------------------------------------------
__global__ __launch_bounds__(128) void attn_flash_kernel(
        const _Float16* __restrict__ Qh, const _Float16* __restrict__ Kh,
        const _Float16* __restrict__ Vt, float* __restrict__ Out) {
    __shared__ __align__(32) _Float16 ldsK[2][32 * DIM];  // 2 x 4KB: K rows
    __shared__ __align__(32) _Float16 ldsV[2][DIM * 32];  // 2 x 4KB: V^T rows
    __shared__ __align__(32) _Float16 ldsP[4][512];       // per-wave P stage

    const int lane = threadIdx.x & 31;
    const int wave = threadIdx.x >> 5;
    const int r0   = (blockIdx.x * 4 + wave) * 16;
    const int n    = lane & 15;
    const int b    = lane >> 4;
    _Float16* wl = ldsP[wave];

    // --- Q fragments (A layout, 2 fragments cover D=64)
    v16h qf[2];
    for (int f = 0; f < 2; ++f) {
        const _Float16* qp = Qh + (size_t)(r0 + n) * DIM + 32 * f;
        v8h lo = *(const v8h*)(qp + 16 * b);
        v8h hi = *(const v8h*)(qp + 16 * b + 8);
        for (int i = 0; i < 8; ++i) { qf[f][i] = lo[i]; qf[f][8 + i] = hi[i]; }
    }

    v8f o[4];
    float mrow[8], lrow[8];
    for (int j = 0; j < 4; ++j)
        for (int v = 0; v < 8; ++v) o[j][v] = 0.f;
    for (int v = 0; v < 8; ++v) { mrow[v] = -3.0e38f; lrow[v] = 0.f; }

    const float kscale = 0.125f * 1.4426950408889634f;  // 1/sqrt(64) * log2(e)

    // Prologue: TDM-load tile 0 into buffer 0.
    if (wave == 0) {
        tdm_load_2d((unsigned)(uintptr_t)(void*)&ldsK[0][0], Kh,
                    DIM, SEQ, DIM, 32, DIM);            // 32 K-rows of 64 f16
        tdm_load_2d((unsigned)(uintptr_t)(void*)&ldsV[0][0], Vt,
                    SEQ, DIM, 32, DIM, SEQ);            // 64 V^T-rows of 32 f16
    }

    for (int k0 = 0; k0 < SEQ; k0 += 32) {
        const int buf = (k0 >> 5) & 1;
        if (wave == 0) __builtin_amdgcn_s_wait_tensorcnt(0);  // tile k0 landed
        asm volatile("s_wait_dscnt 0x0" ::: "memory");        // prior DS reads done
        __syncthreads();
        if (wave == 0 && k0 + 32 < SEQ) {                     // prefetch next tile
            tdm_load_2d((unsigned)(uintptr_t)(void*)&ldsK[buf ^ 1][0],
                        Kh + (size_t)(k0 + 32) * DIM, DIM, SEQ, DIM, 32, DIM);
            tdm_load_2d((unsigned)(uintptr_t)(void*)&ldsV[buf ^ 1][0],
                        Vt + (k0 + 32), SEQ, DIM, 32, DIM, SEQ);
        }
        const _Float16* lk = ldsK[buf];
        const _Float16* lv = ldsV[buf];

        // ---- S = Q @ K^T for two 16-key tiles (B fragments from LDS)
        v8f s[2];
        for (int t = 0; t < 2; ++t) {
            const _Float16* kp = lk + (16 * t + n) * DIM + 16 * b;
            v16h kb0 = *(const v16h*)kp;          // D = 16b + h
            v16h kb1 = *(const v16h*)(kp + 32);   // D = 32 + 16b + h
            v8f z;
            for (int v = 0; v < 8; ++v) z[v] = 0.f;
            z    = wmma_f16(qf[0], kb0, z);
            s[t] = wmma_f16(qf[1], kb1, z);
        }

        // ---- online softmax (row = slot v within a 16-lane half)
        float mnew[8], rsum[8];
        for (int v = 0; v < 8; ++v) {
            float x0 = s[0][v] * kscale;
            float x1 = s[1][v] * kscale;
            float mx = fmaxf(x0, x1);
            mx = fmaxf(mx, __shfl_xor(mx, 1, 32));
            mx = fmaxf(mx, __shfl_xor(mx, 2, 32));
            mx = fmaxf(mx, __shfl_xor(mx, 4, 32));
            mx = fmaxf(mx, __shfl_xor(mx, 8, 32));
            mnew[v] = fmaxf(mrow[v], mx);
            float p0 = exp2f(x0 - mnew[v]);
            float p1 = exp2f(x1 - mnew[v]);
            s[0][v] = p0;
            s[1][v] = p1;
            float rs = p0 + p1;
            rs += __shfl_xor(rs, 1, 32);
            rs += __shfl_xor(rs, 2, 32);
            rs += __shfl_xor(rs, 4, 32);
            rs += __shfl_xor(rs, 8, 32);
            rsum[v] = rs;
        }
        for (int v = 0; v < 8; ++v) {
            float r = exp2f(mrow[v] - mnew[v]);
            lrow[v] = lrow[v] * r + rsum[v];
            mrow[v] = mnew[v];
            for (int j = 0; j < 4; ++j) o[j][v] *= r;
        }

        // ---- transpose P (C layout) -> A-fragment layout through LDS
        const int dlBase = 8 * (lane >> 4);
        for (int v = 0; v < 8; ++v) {
            int dl = v + dlBase;
            wl[dl * 16 + (lane & 15)]        = (_Float16)s[0][v];
            wl[(dl + 16) * 16 + (lane & 15)] = (_Float16)s[1][v];
        }
        asm volatile("s_wait_dscnt 0x0" ::: "memory");
        v16h pf = *(const v16h*)(wl + lane * 16);

        // ---- O += P @ V (B fragments from LDS: 32 contiguous keys per row)
        for (int j = 0; j < 4; ++j) {
            const _Float16* vp = lv + (16 * j + n) * 32 + 16 * b;
            v16h vb = *(const v16h*)vp;           // key = 16b + h
            o[j] = wmma_f16(pf, vb, o[j]);
        }
    }

    // ---- normalize and write f32 output [8192][64]
    for (int v = 0; v < 8; ++v) {
        float inv = 1.0f / lrow[v];
        int row = r0 + v + 8 * b;
        for (int j = 0; j < 4; ++j)
            Out[(size_t)row * DIM + 16 * j + n] = o[j][v] * inv;
    }
}

// ---------------------------------------------------------------------------
extern "C" void kernel_launch(void* const* d_in, const int* in_sizes, int n_in,
                              void* d_out, int out_size, void* d_ws, size_t ws_size,
                              hipStream_t stream) {
    const float* X  = (const float*)d_in[0];
    const float* WQ = (const float*)d_in[1];
    const float* WK = (const float*)d_in[2];
    const float* WV = (const float*)d_in[3];
    float* Out = (float*)d_out;

    char* ws = (char*)d_ws;
    _Float16* Qh = (_Float16*)(ws);                // 1 MB
    _Float16* Kh = (_Float16*)(ws + (1u << 20));   // 1 MB
    _Float16* Vt = (_Float16*)(ws + (2u << 20));   // 1 MB (transposed V)
    _Float16* Wt = (_Float16*)(ws + (3u << 20));   // 288 KB

    attn_wt_kernel<<<(3 * DIM * EMB + 255) / 256, 256, 0, stream>>>(WQ, WK, WV, Wt);

    attn_proj_kernel<<<SEQ / 16 / 2, 64, 0, stream>>>(X, Wt, Qh, Kh, Vt);

    attn_flash_kernel<<<SEQ / (16 * 4), 128, 0, stream>>>(Qh, Kh, Vt, Out);
}
